// Reservoir_9646496546838
// MI455X (gfx1250) — matrix operand order repaired and checked by reading, not compiled, not run on hardware
//
#include <hip/hip_runtime.h>
#include <hip/hip_bf16.h>
#include <math.h>

// ---------------------------------------------------------------------------
// Echo-state reservoir for MI455X (gfx1250, wave32).
//   proj = x @ W_in.T : bf16 WMMA GEMM (8192x64x2048) -> d_out
//   scan: s = (1-a)s + a*tanh(proj[t] + s @ W.T)
//     persistent kernel: 64 blocks x 8 waves = 512 waves.
//     Each block owns 2 row-tiles (16 rows each) with 4-way split-K per tile;
//     each wave streams 16KB of bf16 W per step (L2-resident, 8MB total) and
//     runs two independent v_wmma_f32_16x16x32_bf16 accumulation chains.
//     Partials reduced via LDS; one device-wide barrier per step.
// ---------------------------------------------------------------------------

typedef __attribute__((ext_vector_type(16))) __bf16 v16bf;
typedef __attribute__((ext_vector_type(8)))  __bf16 v8bf;
typedef __attribute__((ext_vector_type(8)))  float  v8f;

#define N_RES   2048
#define I_IN    64
#define T_STEPS 8192
#define LEAK    0.1f

// ---- workspace layout (64B-aligned offsets) -------------------------------
#define WS_CTR   0                        // 2 x u32 (arrive, gen)
#define WS_SF    64                       // 2 x 2048 f32 state ping-pong
#define WS_SB    (WS_SF + 2*N_RES*4)      // 2 x 2048 bf16 state ping-pong
#define WS_WB    24640                    // 2048x2048 bf16 W
#define WS_WINB  (WS_WB + N_RES*N_RES*2)  // 2048x64 bf16 W_in
#define WS_XB    (WS_WINB + N_RES*I_IN*2) // 8192x64 bf16 x

// ---------------------------------------------------------------------------
// Prep: fp32 -> bf16 conversions, state init, barrier counter reset.
// Runs every launch -> deterministic across graph replays.
// ---------------------------------------------------------------------------
__global__ void prep_kernel(const float* __restrict__ x,
                            const float* __restrict__ s0,
                            const float* __restrict__ Win,
                            const float* __restrict__ W,
                            __bf16* __restrict__ Wb,
                            __bf16* __restrict__ Winb,
                            __bf16* __restrict__ xb,
                            float*  __restrict__ sf,
                            __bf16* __restrict__ sb,
                            unsigned* __restrict__ ctr) {
  long i = (long)blockIdx.x * blockDim.x + threadIdx.x;
  if (i < (long)N_RES * N_RES) Wb[i]   = (__bf16)W[i];
  if (i < N_RES * I_IN)        Winb[i] = (__bf16)Win[i];
  if (i < T_STEPS * I_IN)      xb[i]   = (__bf16)x[i];
  if (i < N_RES) { float s = s0[i]; sf[i] = s; sb[i] = (__bf16)s; }
  if (i < 2) ctr[i] = 0u;
}

// ---------------------------------------------------------------------------
// A-tile loader: 16x32 bf16 tile, lanes 0-15 hold K 0..7 / 16..23,
// lanes 16-31 hold K 8..15 / 24..31 (two 16B loads per lane).
// ---------------------------------------------------------------------------
__device__ __forceinline__ v16bf load_a_tile(const __bf16* p) {
  v8bf lo = *(const v8bf*)(p);
  v8bf hi = *(const v8bf*)(p + 16);
  v16bf a;
#pragma unroll
  for (int i = 0; i < 8; ++i) { a[i] = lo[i]; a[8 + i] = hi[i]; }
  return a;
}

// ---------------------------------------------------------------------------
// proj = x @ W_in.T. One 16x16 output tile per wave, K=64 -> 2 WMMAs.
// ---------------------------------------------------------------------------
__global__ void __launch_bounds__(256)
proj_kernel(const __bf16* __restrict__ xb,
            const __bf16* __restrict__ Winb,
            float* __restrict__ out) {
  const int lane = threadIdx.x & 31;
  const int wave = threadIdx.x >> 5;
  const int tile = blockIdx.x * 8 + wave;   // 0..65535
  const int mt = tile >> 7;                 // 512 M-tiles
  const int nt = tile & 127;                // 128 N-tiles
  const int arow = mt * 16 + (lane & 15);
  const int bcol = nt * 16 + (lane & 15);
  const int aoff = (lane < 16) ? 0 : 8;
  const int boff = (lane < 16) ? 0 : 8;

  v8f acc = {};
#pragma unroll
  for (int kt = 0; kt < 2; ++kt) {
    const int k0 = kt * 32;
    v16bf a = load_a_tile(xb + (size_t)arow * I_IN + k0 + aoff);
    v16bf b = load_a_tile(Winb + (size_t)bcol * I_IN + k0 + boff);
    acc = __builtin_amdgcn_wmma_f32_16x16x32_bf16(
        false, a, false, b, (short)0, acc, false, false);
  }

  const int half = (lane >= 16);
  float* orow = out + (size_t)(mt * 16 + half * 8) * N_RES + nt * 16 + (lane & 15);
#pragma unroll
  for (int v = 0; v < 8; ++v) orow[(size_t)v * N_RES] = acc[v];
}

// ---------------------------------------------------------------------------
// Sense-reversal grid barrier (agent-scope atomics in workspace).
// ---------------------------------------------------------------------------
__device__ __forceinline__ void grid_barrier(unsigned* arrive, unsigned* gen,
                                             unsigned nblocks, unsigned g) {
  __syncthreads();
  if (threadIdx.x == 0) {
    __threadfence();  // release prior state writes to device scope
    unsigned a = __hip_atomic_fetch_add(arrive, 1u, __ATOMIC_ACQ_REL,
                                        __HIP_MEMORY_SCOPE_AGENT);
    if (a == nblocks - 1u) {
      __hip_atomic_store(arrive, 0u, __ATOMIC_RELAXED, __HIP_MEMORY_SCOPE_AGENT);
      __hip_atomic_store(gen, g + 1u, __ATOMIC_RELEASE, __HIP_MEMORY_SCOPE_AGENT);
    } else {
      while (__hip_atomic_load(gen, __ATOMIC_ACQUIRE,
                               __HIP_MEMORY_SCOPE_AGENT) <= g)
        __builtin_amdgcn_s_sleep(2);
    }
  }
  __syncthreads();
}

// ---------------------------------------------------------------------------
// Persistent recurrence. Block = 256 threads (8 waves): 2 row-tiles x 4
// K-splits. Wave (tl, sp) accumulates rows [.., ..+16) x K [sp*512, +512)
// with two interleaved WMMA chains; lanes 0/16 dump 8 partials each to LDS;
// first wave reduces, applies tanh + leaky blend, stores state + output.
// ---------------------------------------------------------------------------
__global__ void __launch_bounds__(256)
reservoir_kernel(const __bf16* __restrict__ Wb,
                 float* __restrict__ out,     // [T,N]: proj in, states out
                 float* __restrict__ sf,      // 2 x N f32 state ping-pong
                 __bf16* __restrict__ sb,     // 2 x N bf16 state ping-pong
                 unsigned* __restrict__ ctr) {
  __shared__ float part[2][4][16];            // [tile_local][split][row]

  const int lane = threadIdx.x & 31;
  const int wave = threadIdx.x >> 5;          // 0..7
  const int tl   = wave >> 2;                 // tile within block: 0..1
  const int sp   = wave & 3;                  // K-split: 0..3
  const int tile = blockIdx.x * 2 + tl;       // 0..127
  const int rbase = tile * 16;
  const int arow = rbase + (lane & 15);
  const int aoff = (lane < 16) ? 0 : 8;
  const int boff = (lane < 16) ? 0 : 8;
  const int half = (lane >= 16);
  const int kbase = sp * (N_RES / 4);         // 512 K per split = 16 k-tiles
  const __bf16* __restrict__ wrow = Wb + (size_t)arow * N_RES + kbase;

  unsigned* arrive = ctr;
  unsigned* gen    = ctr + 1;
  unsigned g = 0;

  for (int t = 0; t < T_STEPS; ++t) {
    const int p = t & 1;
    const __bf16* sbi = sb + p * N_RES + kbase;
    __bf16*       sbo = sb + (1 - p) * N_RES;
    const float*  sfi = sf + p * N_RES;
    float*        sfo = sf + (1 - p) * N_RES;

    // Two independent accumulation chains over 16 k-tiles (16KB of W).
    v8f acc0 = {}, acc1 = {};
#pragma unroll 2
    for (int kt = 0; kt < 16; kt += 2) {
      const int k0 = kt * 32;
      __builtin_prefetch((const void*)(wrow + k0 + 512 + aoff), 0, 0);
      v16bf a0 = load_a_tile(wrow + k0 + aoff);
      v16bf b0 = load_a_tile(sbi + k0 + boff);
      v16bf a1 = load_a_tile(wrow + k0 + 32 + aoff);
      v16bf b1 = load_a_tile(sbi + k0 + 32 + boff);
      acc0 = __builtin_amdgcn_wmma_f32_16x16x32_bf16(
          false, a0, false, b0, (short)0, acc0, false, false);
      acc1 = __builtin_amdgcn_wmma_f32_16x16x32_bf16(
          false, a1, false, b1, (short)0, acc1, false, false);
    }

    // Lanes 0 and 16 hold rows half*8+v in VGPR v of the D tile.
    if ((lane & 15) == 0) {
#pragma unroll
      for (int v = 0; v < 8; ++v)
        part[tl][sp][half * 8 + v] = acc0[v] + acc1[v];
    }
    __syncthreads();

    // First wave: one thread per row (2 tiles x 16 rows = 32 rows).
    if (threadIdx.x < 32) {
      const int rtl = threadIdx.x >> 4;
      const int rr  = threadIdx.x & 15;
      const int row = (blockIdx.x * 2 + rtl) * 16 + rr;
      float y = part[rtl][0][rr] + part[rtl][1][rr] +
                part[rtl][2][rr] + part[rtl][3][rr];
      float* orow = out + (size_t)t * N_RES;
      const float sn = tanhf(orow[row] + y);
      const float s  = (1.0f - LEAK) * sfi[row] + LEAK * sn;
      orow[row] = s;
      sfo[row]  = s;
      sbo[row]  = (__bf16)s;
    }

    if (t + 1 < T_STEPS) { grid_barrier(arrive, gen, gridDim.x, g); ++g; }
    else break;
  }
}

// ---------------------------------------------------------------------------
extern "C" void kernel_launch(void* const* d_in, const int* in_sizes, int n_in,
                              void* d_out, int out_size, void* d_ws, size_t ws_size,
                              hipStream_t stream) {
  const float* x   = (const float*)d_in[0];   // [T, I]
  const float* s0  = (const float*)d_in[1];   // [N]
  const float* Win = (const float*)d_in[2];   // [N, I]
  const float* W   = (const float*)d_in[3];   // [N, N]
  float* out = (float*)d_out;                 // [T, N]

  char* ws = (char*)d_ws;
  unsigned* ctr  = (unsigned*)(ws + WS_CTR);
  float*    sf   = (float*)   (ws + WS_SF);
  __bf16*   sb   = (__bf16*)  (ws + WS_SB);
  __bf16*   Wb   = (__bf16*)  (ws + WS_WB);
  __bf16*   Winb = (__bf16*)  (ws + WS_WINB);
  __bf16*   xb   = (__bf16*)  (ws + WS_XB);

  const int prep_n = N_RES * N_RES;  // largest conversion
  prep_kernel<<<(prep_n + 255) / 256, 256, 0, stream>>>(
      x, s0, Win, W, Wb, Winb, xb, sf, sb, ctr);

  // 512 M-tiles x 128 N-tiles = 65536 tiles, 8 waves/block -> 8192 blocks
  proj_kernel<<<8192, 256, 0, stream>>>(xb, Winb, out);

  // 64 blocks x 8 waves = 512 waves: 128 row tiles x 4-way split-K
  reservoir_kernel<<<64, 256, 0, stream>>>(Wb, out, sf, sb, ctr);
}
